// Flux2ParallelSelfAttention_16887811408265
// MI455X (gfx1250) — compile-verified, matrix operand order checked by reading, not run on hardware
//
#include <hip/hip_runtime.h>
#include <hip/hip_bf16.h>

typedef __bf16 bf16_t;
typedef bf16_t v16bf __attribute__((ext_vector_type(16)));
typedef float  v8f   __attribute__((ext_vector_type(8)));
typedef unsigned int u32x4 __attribute__((ext_vector_type(4)));
typedef int          i32x4 __attribute__((ext_vector_type(4)));
typedef int          i32x8 __attribute__((ext_vector_type(8)));

#define HEADS     24
#define HEAD_DIM  128
#define INNER     (HEADS * HEAD_DIM)        // 3072
#define MLP_HID   12288
#define FUSED     (3 * INNER + 2 * MLP_HID) // 33792
#define SEQ       2048
#define DMODEL    3072
#define CAT       (INNER + MLP_HID)         // 15360
#define RSCALE    0.08838834764831845f      // 1/sqrt(128)

__device__ __forceinline__ bf16_t f2bf(float f) {
  unsigned u = __builtin_bit_cast(unsigned, f);
  u += 0x7FFFu + ((u >> 16) & 1u);                 // round-to-nearest-even
  unsigned short h = (unsigned short)(u >> 16);
  return __builtin_bit_cast(bf16_t, h);
}
__device__ __forceinline__ float bf2f(bf16_t h) {
  unsigned short s = __builtin_bit_cast(unsigned short, h);
  unsigned u = ((unsigned)s) << 16;
  return __builtin_bit_cast(float, u);
}

__device__ __forceinline__ unsigned lds_offset_of(const void* p) {
  // flat LDS-aperture addresses carry the LDS byte offset in the low 32 bits
  return (unsigned)(uintptr_t)p;
}

// ---------------------------------------------------------------------------
// Tensor Data Mover: 2D bf16 tile (tile_d1 rows x tile_d0 elements, row
// stride stride_elts) global -> LDS. D# per CDNA5 ISA 8.3/8.4. Wave-level,
// tracked with TENSORcnt.
// ---------------------------------------------------------------------------
__device__ __forceinline__ void tdm_load_2d(unsigned lds_off, const void* gaddr,
                                            unsigned tile_d0, unsigned tile_d1,
                                            unsigned long long stride_elts) {
  unsigned long long ga = (unsigned long long)(uintptr_t)gaddr;
  u32x4 g0;
  g0[0] = 1u;                                          // count=1, user D#
  g0[1] = lds_off;                                     // lds_addr (bytes)
  g0[2] = (unsigned)(ga & 0xFFFFFFFFu);                // global_addr[31:0]
  g0[3] = (unsigned)((ga >> 32) & 0x01FFFFFFu)         // global_addr[56:32]
        | 0x80000000u;                                 // type=2 ("image")
  i32x8 g1;
  g1[0] = (int)(1u << 16);                             // data_size=1 -> 2 bytes
  g1[1] = (int)((tile_d0 & 0xFFFFu) << 16);            // tensor_dim0[15:0]
  g1[2] = (int)(((tile_d0 >> 16) & 0xFFFFu)            // tensor_dim0[31:16]
        | ((tile_d1 & 0xFFFFu) << 16));                // tensor_dim1[15:0]
  g1[3] = (int)(((tile_d1 >> 16) & 0xFFFFu)            // tensor_dim1[31:16]
        | ((tile_d0 & 0xFFFFu) << 16));                // tile_dim0
  g1[4] = (int)(tile_d1 & 0xFFFFu);                    // tile_dim1 (tile_dim2=0)
  g1[5] = (int)(stride_elts & 0xFFFFFFFFull);          // tensor_dim0_stride lo
  g1[6] = (int)((stride_elts >> 32) & 0xFFFFull);      // tensor_dim0_stride hi
  g1[7] = 0;
  i32x4 gz = {};
#if defined(__clang_major__) && (__clang_major__ >= 23)
  i32x8 gz8 = {};
  __builtin_amdgcn_tensor_load_to_lds(g0, g1, gz, gz, gz8, 0);
#else
  __builtin_amdgcn_tensor_load_to_lds(g0, g1, gz, gz, 0);
#endif
}

// ---------------------------------------------------------------------------
// Async global -> LDS copy (per-lane 16B), tracked with ASYNCcnt.
// ---------------------------------------------------------------------------
__device__ __forceinline__ void async_load_b128(unsigned lds_byte_off, const void* g) {
  unsigned long long ga = (unsigned long long)(uintptr_t)g;
  asm volatile("global_load_async_to_lds_b128 %0, %1, off"
               :: "v"(lds_byte_off), "v"(ga) : "memory");
}
__device__ __forceinline__ void wait_asynccnt0() {
  asm volatile("s_wait_asynccnt 0x0" ::: "memory");
}

// ---------------------------------------------------------------------------
// Tiled GEMM: C[M,N] = A[M,K] * B[K,N] (+bias). Double-buffered LDS pipeline:
// stage tile kt+1 (fp32 load+convert, or TDM DMA for bf16 A) while WMMAs run
// on tile kt; one barrier per iteration. v_wmma_f32_16x16x32_bf16.
// Block = 128x128 tile, 256 threads = 8 waves (wave32), wave tile = 64x32.
// ---------------------------------------------------------------------------
template <bool A_IS_F32, bool OUT_IS_BF16>
__global__ __launch_bounds__(256) void gemm_wmma(
    const void* __restrict__ Ap, int lda,
    const float* __restrict__ B, int ldb,
    void* __restrict__ Cp, int ldc,
    const float* __restrict__ bias,
    int K)
{
  __shared__ __align__(32) bf16_t As[2][128 * 32];   // row-major [m][k]
  __shared__ __align__(32) bf16_t Bs[2][128 * 32];   // n-major  [n][k]

  const int tid  = threadIdx.x;
  const int lane = tid & 31;
  const int w    = tid >> 5;   // 0..7
  const int wr   = w >> 2;     // 0..1 : 64-row strip
  const int wc   = w & 3;      // 0..3 : 32-col strip
  const int m0   = blockIdx.y * 128;
  const int n0   = blockIdx.x * 128;
  const int nk   = K >> 5;

  auto stage_a = [&](int kt, bf16_t* dst) {
    if (A_IS_F32) {
      const float* A = (const float*)Ap;
      for (int i = tid; i < 128 * 8; i += 256) {
        int r = i >> 3, c4 = (i & 7) << 2;
        const float4 f =
            *reinterpret_cast<const float4*>(&A[(size_t)(m0 + r) * lda + kt * 32 + c4]);
        dst[r * 32 + c4 + 0] = f2bf(f.x);
        dst[r * 32 + c4 + 1] = f2bf(f.y);
        dst[r * 32 + c4 + 2] = f2bf(f.z);
        dst[r * 32 + c4 + 3] = f2bf(f.w);
      }
    } else {
      if (w == 0) {   // Tensor Data Mover DMA, runs behind the WMMA burst
        const bf16_t* A = (const bf16_t*)Ap;
        tdm_load_2d(lds_offset_of(dst), &A[(size_t)m0 * lda + kt * 32],
                    /*tile_d0=*/32, /*tile_d1=*/128, (unsigned long long)lda);
      }
    }
  };
  auto stage_b = [&](int kt, bf16_t* dst) {
    for (int i = tid; i < 32 * 32; i += 256) {
      int k = i >> 5, nn = (i & 31) << 2;
      const float4 f =
          *reinterpret_cast<const float4*>(&B[(size_t)(kt * 32 + k) * ldb + n0 + nn]);
      dst[(nn + 0) * 32 + k] = f2bf(f.x);
      dst[(nn + 1) * 32 + k] = f2bf(f.y);
      dst[(nn + 2) * 32 + k] = f2bf(f.z);
      dst[(nn + 3) * 32 + k] = f2bf(f.w);
    }
  };

  const v8f vzero = {};
  v8f acc[4][2];
#pragma unroll
  for (int i = 0; i < 4; ++i)
#pragma unroll
    for (int j = 0; j < 2; ++j) acc[i][j] = vzero;

  // prologue: stage tile 0
  stage_a(0, As[0]);
  stage_b(0, Bs[0]);
  if (!A_IS_F32 && w == 0) __builtin_amdgcn_s_wait_tensorcnt(0);
  __syncthreads();

  for (int kt = 0; kt < nk; ++kt) {
    const int cur = kt & 1, nxt = cur ^ 1;
    if (kt + 1 < nk) {                 // stage next tile into alternate buffer
      stage_a(kt + 1, As[nxt]);
      stage_b(kt + 1, Bs[nxt]);
    }
    if (kt + 2 < nk)                   // gfx1250 global_prefetch_b8
      __builtin_prefetch(&B[(size_t)((kt + 2) * 32) * ldb + n0], 0, 1);

    // ---- 8 WMMAs per wave on the current buffer ----
    v16bf a[4], b[2];
#pragma unroll
    for (int i = 0; i < 4; ++i) {
      int row = wr * 64 + i * 16 + (lane & 15);
      a[i] = *reinterpret_cast<const v16bf*>(&As[cur][row * 32 + (lane >> 4) * 16]);
    }
#pragma unroll
    for (int j = 0; j < 2; ++j) {
      int col = wc * 32 + j * 16 + (lane & 15);
      b[j] = *reinterpret_cast<const v16bf*>(&Bs[cur][col * 32 + (lane >> 4) * 16]);
    }
#pragma unroll
    for (int i = 0; i < 4; ++i)
#pragma unroll
      for (int j = 0; j < 2; ++j)
        acc[i][j] = __builtin_amdgcn_wmma_f32_16x16x32_bf16(
            false, a[i], false, b[j], (short)0, acc[i][j], false, false);

    if (!A_IS_F32 && w == 0 && kt + 1 < nk)
      __builtin_amdgcn_s_wait_tensorcnt(0);   // TDM of nxt done before barrier
    __syncthreads();
  }

  // ---- epilogue: 16x16 f32 C layout: n = lane%16, m = vgpr + 8*(lane/16) ----
#pragma unroll
  for (int i = 0; i < 4; ++i) {
#pragma unroll
    for (int j = 0; j < 2; ++j) {
      int mb = m0 + wr * 64 + i * 16 + (lane >> 4) * 8;
      int nb = n0 + wc * 32 + j * 16 + (lane & 15);
#pragma unroll
      for (int v = 0; v < 8; ++v) {
        float r = acc[i][j][v];
        size_t idx = (size_t)(mb + v) * ldc + nb;
        if (OUT_IS_BF16)
          ((bf16_t*)Cp)[idx] = f2bf(r);
        else
          ((float*)Cp)[idx] = r + bias[nb];
      }
    }
  }
}

// ---------------------------------------------------------------------------
// RMSNorm + RoPE on q/k, in place in the bf16 h buffer.
// One wave per (token, head, q|k); 128 dims -> 4 per lane (wave32).
// ---------------------------------------------------------------------------
__global__ __launch_bounds__(32) void rmsnorm_rope(
    bf16_t* __restrict__ h, const float* __restrict__ cosb,
    const float* __restrict__ sinb, const float* __restrict__ wq,
    const float* __restrict__ wk)
{
  const int s    = blockIdx.x;
  const int hd   = blockIdx.y;
  const int qk   = blockIdx.z;       // 0 = q, 1 = k
  const int lane = threadIdx.x;
  const float* w = qk ? wk : wq;
  bf16_t* p = h + (size_t)s * FUSED + qk * INNER + hd * HEAD_DIM + lane * 4;

  float x0 = bf2f(p[0]), x1 = bf2f(p[1]), x2 = bf2f(p[2]), x3 = bf2f(p[3]);
  float ss = x0 * x0 + x1 * x1 + x2 * x2 + x3 * x3;
  for (int m = 16; m >= 1; m >>= 1) ss += __shfl_xor(ss, m, 32);  // wave32
  float r = rsqrtf(ss * (1.0f / HEAD_DIM) + 1e-5f);

  const int d0 = lane * 4;
  x0 *= r * w[d0 + 0]; x1 *= r * w[d0 + 1];
  x2 *= r * w[d0 + 2]; x3 *= r * w[d0 + 3];

  const int pr = lane * 2;
  float c0 = cosb[s * 64 + pr],     s0 = sinb[s * 64 + pr];
  float c1 = cosb[s * 64 + pr + 1], s1 = sinb[s * 64 + pr + 1];
  p[0] = f2bf(x0 * c0 - x1 * s0);
  p[1] = f2bf(x0 * s0 + x1 * c0);
  p[2] = f2bf(x2 * c1 - x3 * s1);
  p[3] = f2bf(x2 * s1 + x3 * c1);
}

// ---------------------------------------------------------------------------
// SiLU(x1) * x2 into the concat buffer (cols INNER..CAT-1).
// ---------------------------------------------------------------------------
__global__ __launch_bounds__(256) void silu_mul(
    const bf16_t* __restrict__ h, bf16_t* __restrict__ a2)
{
  const size_t n = (size_t)SEQ * MLP_HID;
  for (size_t i = (size_t)blockIdx.x * 256 + threadIdx.x; i < n;
       i += (size_t)gridDim.x * 256) {
    size_t row = i / MLP_HID;
    int j = (int)(i % MLP_HID);
    float x1 = bf2f(h[row * FUSED + 3 * INNER + j]);
    float x2 = bf2f(h[row * FUSED + 3 * INNER + MLP_HID + j]);
    float sl = x1 / (1.0f + __expf(-x1));
    a2[row * CAT + INNER + j] = f2bf(sl * x2);
  }
}

// ---------------------------------------------------------------------------
// Flash attention: one block = (head, 32 q rows), 128 threads = 4 waves.
// Q tile: async-to-LDS copies (ASYNCcnt). K/V tiles: TDM DMA (TENSORcnt).
// QK^T and PV on v_wmma_f32_16x16x32_bf16; online softmax in LDS.
// ---------------------------------------------------------------------------
__global__ __launch_bounds__(128) void flash_attn(
    const bf16_t* __restrict__ h, bf16_t* __restrict__ a2)
{
  __shared__ __align__(32) bf16_t Qs[32 * 128];  // [qrow][d]
  __shared__ __align__(32) bf16_t Ks[32 * 128];  // [krow][d]
  __shared__ __align__(32) bf16_t Vs[32 * 128];  // [krow][d] TDM staging
  __shared__ __align__(32) bf16_t Vt[128 * 32];  // [d][krow] (transposed)
  __shared__ __align__(32) float  Sc[32 * 32];   // scores
  __shared__ __align__(32) bf16_t Pt[32 * 32];   // exp(scores)
  __shared__ __align__(32) float  Os[32 * 128];  // running output
  __shared__ float mrow[32], lrow[32], arow[32];

  const int tid  = threadIdx.x;
  const int lane = tid & 31;
  const int w    = tid >> 5;   // 0..3
  const int q0   = blockIdx.x * 32;
  const int head = blockIdx.y;

  // async DMA of the Q tile: 512 x 16B chunks across 128 lanes
  {
    const unsigned qoff = lds_offset_of(Qs);
    for (int i = tid; i < 512; i += 128) {
      int r = i >> 4, c8 = (i & 15) << 3;
      async_load_b128(qoff + (unsigned)(r * 128 + c8) * 2,
                      &h[(size_t)(q0 + r) * FUSED + head * HEAD_DIM + c8]);
    }
    wait_asynccnt0();
  }
  for (int i = tid; i < 32 * 128; i += 128) Os[i] = 0.0f;
  if (tid < 32) { mrow[tid] = -1e30f; lrow[tid] = 0.0f; }

  for (int kt = 0; kt < SEQ / 32; ++kt) {
    const int k0 = kt * 32;
    __syncthreads();
    if (w == 0) {   // TDM DMA of K and V tiles for this step
      tdm_load_2d(lds_offset_of(Ks),
                  &h[(size_t)k0 * FUSED + INNER + head * HEAD_DIM],
                  HEAD_DIM, 32, (unsigned long long)FUSED);
      tdm_load_2d(lds_offset_of(Vs),
                  &h[(size_t)k0 * FUSED + 2 * INNER + head * HEAD_DIM],
                  HEAD_DIM, 32, (unsigned long long)FUSED);
      __builtin_amdgcn_s_wait_tensorcnt(0);
    }
    __syncthreads();
    // transpose V in LDS for contiguous B-fragments in PV
    for (int i = tid; i < 32 * 128; i += 128) {
      int r = i >> 7, c = i & 127;
      Vt[c * 32 + r] = Vs[i];
    }
    __syncthreads();

    // S = Q K^T : 4 waves x one 16x16 tile, K-dim 128 = 4 chained wmma
    {
      int tr = w >> 1, tc = w & 1;
      v8f acc = {};
#pragma unroll
      for (int kk = 0; kk < 4; ++kk) {
        v16bf a = *reinterpret_cast<const v16bf*>(
            &Qs[(tr * 16 + (lane & 15)) * 128 + kk * 32 + (lane >> 4) * 16]);
        v16bf b = *reinterpret_cast<const v16bf*>(
            &Ks[(tc * 16 + (lane & 15)) * 128 + kk * 32 + (lane >> 4) * 16]);
        acc = __builtin_amdgcn_wmma_f32_16x16x32_bf16(
            false, a, false, b, (short)0, acc, false, false);
      }
#pragma unroll
      for (int v = 0; v < 8; ++v) {
        int m = tr * 16 + v + (lane >> 4) * 8;
        int n = tc * 16 + (lane & 15);
        Sc[m * 32 + n] = acc[v] * RSCALE;
      }
    }
    __syncthreads();

    // online softmax: one thread per row
    if (tid < 32) {
      int row = tid;
      float rmax = -1e30f;
      for (int j = 0; j < 32; ++j) rmax = fmaxf(rmax, Sc[row * 32 + j]);
      float mnew  = fmaxf(mrow[row], rmax);
      float alpha = __expf(mrow[row] - mnew);
      float lsum  = 0.0f;
      for (int j = 0; j < 32; ++j) {
        float pv = __expf(Sc[row * 32 + j] - mnew);
        Pt[row * 32 + j] = f2bf(pv);
        lsum += pv;
      }
      lrow[row] = lrow[row] * alpha + lsum;
      mrow[row] = mnew;
      arow[row] = alpha;
    }
    __syncthreads();
    for (int i = tid; i < 32 * 128; i += 128) Os[i] *= arow[i >> 7];
    __syncthreads();

    // O += P V : 32x128 out = 2x8 tiles, 4 per wave; K-dim 32 = 1 wmma each
#pragma unroll
    for (int t4 = 0; t4 < 4; ++t4) {
      int t  = w * 4 + t4;
      int tr = t >> 3, tc = t & 7;
      v8f c;
#pragma unroll
      for (int v = 0; v < 8; ++v)
        c[v] = Os[(tr * 16 + v + (lane >> 4) * 8) * 128 + tc * 16 + (lane & 15)];
      v16bf a = *reinterpret_cast<const v16bf*>(
          &Pt[(tr * 16 + (lane & 15)) * 32 + (lane >> 4) * 16]);
      v16bf b = *reinterpret_cast<const v16bf*>(
          &Vt[(tc * 16 + (lane & 15)) * 32 + (lane >> 4) * 16]);
      c = __builtin_amdgcn_wmma_f32_16x16x32_bf16(
          false, a, false, b, (short)0, c, false, false);
#pragma unroll
      for (int v = 0; v < 8; ++v)
        Os[(tr * 16 + v + (lane >> 4) * 8) * 128 + tc * 16 + (lane & 15)] = c[v];
    }
  }
  __syncthreads();
  for (int i = tid; i < 32 * 128; i += 128) {
    int r = i >> 7, c = i & 127;
    a2[(size_t)(q0 + r) * CAT + head * HEAD_DIM + c] = f2bf(Os[i] / lrow[r]);
  }
}

// ---------------------------------------------------------------------------
extern "C" void kernel_launch(void* const* d_in, const int* in_sizes, int n_in,
                              void* d_out, int out_size, void* d_ws, size_t ws_size,
                              hipStream_t stream) {
  const float* hs    = (const float*)d_in[0];  // hidden_states (2048x3072)
  const float* cosb  = (const float*)d_in[1];  // (2048x64)
  const float* sinb  = (const float*)d_in[2];  // (2048x64)
  const float* wqkv  = (const float*)d_in[3];  // (3072x33792)
  const float* wout  = (const float*)d_in[4];  // (15360x3072)
  const float* bout  = (const float*)d_in[5];  // (3072)
  const float* nqw   = (const float*)d_in[6];  // (128)
  const float* nkw   = (const float*)d_in[7];  // (128)
  float* out = (float*)d_out;

  bf16_t* hbuf = (bf16_t*)d_ws;                         // 2048 x 33792 bf16
  bf16_t* a2   = hbuf + (size_t)SEQ * FUSED;            // 2048 x 15360 bf16

  // 1) fused QKV + MLP projection
  gemm_wmma<true, true><<<dim3(FUSED / 128, SEQ / 128), 256, 0, stream>>>(
      hs, DMODEL, wqkv, FUSED, hbuf, FUSED, nullptr, DMODEL);
  // 2) RMSNorm + RoPE on q, k (in place)
  rmsnorm_rope<<<dim3(SEQ, HEADS, 2), 32, 0, stream>>>(hbuf, cosb, sinb, nqw, nkw);
  // 3) SiLU-gated MLP into concat buffer
  silu_mul<<<dim3(1024), 256, 0, stream>>>(hbuf, a2);
  // 4) flash attention into concat buffer
  flash_attn<<<dim3(SEQ / 32, HEADS), 128, 0, stream>>>(hbuf, a2);
  // 5) output projection + bias
  gemm_wmma<false, false><<<dim3(DMODEL / 128, SEQ / 128), 256, 0, stream>>>(
      a2, CAT, wout, DMODEL, out, DMODEL, bout, CAT);
}
